// RelativeCrossAttentionLayer_67731634258348
// MI455X (gfx1250) — matrix operand order repaired and checked by reading, not compiled, not run on hardware
//
#include <hip/hip_runtime.h>
#include <stdint.h>

#define E_DIM  1024
#define H_DIM  8
#define HD_DIM 128
#define T_DIM  512
#define S_DIM  2048
#define B_DIM  8
#define LN_EPS 1e-5f

typedef __attribute__((ext_vector_type(16))) __bf16 v16bf;
typedef __attribute__((ext_vector_type(8)))  __bf16 v8bf;
typedef __attribute__((ext_vector_type(8)))  float  v8f;
typedef __attribute__((ext_vector_type(4)))  float  v4f;

union ABfrag { v16bf v; v8bf h[2]; };

// ---------------- fp32 -> bf16 conversion ----------------
__global__ void cvt_f32_to_bf16(const float* __restrict__ src,
                                __bf16* __restrict__ dst, long n) {
    long i = ((long)blockIdx.x * blockDim.x + threadIdx.x) * 4;
    if (i + 3 < n) {
        v4f x = *(const v4f*)(src + i);
        dst[i + 0] = (__bf16)x.x;
        dst[i + 1] = (__bf16)x.y;
        dst[i + 2] = (__bf16)x.z;
        dst[i + 3] = (__bf16)x.w;
    }
}

// ---------------- generic bf16 WMMA GEMM:  C = A @ W^T (+bias) ----------------
// One wave per block computes a 64x64 macro-tile = 4x4 WMMA tiles (register blocked:
// 16 v8f accumulators, 4 A-fragments + 4 B-fragments reused per K-step -> 16 WMMA
// per 8 b128-load clauses, 32 FLOP/byte of fragment traffic).
// A: (M x K) bf16, row-major, leading dim lda, per-z stride strideAz
// W: (N x K) bf16, row-major (K contiguous), leading dim ldw, per-z stride strideWz
// MODE 0: out = fp32 row-major (m*ldc + n)
// MODE 1: out = bf16 v-transposed: m=(s*B+b), n=(h*HD+d) -> vt[((b*H+h)*HD+d)*S + s]
// MODE 2: out = bf16 row-major per z: out[z*strideOz + m*ldc + n]
// MODE 3: out = bf16 attn rows: z=(b*H+h), out[(m*B + b)*E + h*HD + n]
template <int MODE>
__global__ __launch_bounds__(32)
void wmma_gemm_kernel(const __bf16* __restrict__ A, long lda, long strideAz,
                      const __bf16* __restrict__ W, long ldw, long strideWz,
                      const float* __restrict__ bias,
                      void* __restrict__ outp, long ldc, long strideOz,
                      int K) {
    const int lane = threadIdx.x & 31;
    const int half = lane >> 4;           // 0: lanes 0-15, 1: lanes 16-31
    const int l16  = lane & 15;
    const int m0 = blockIdx.y * 64;
    const int n0 = blockIdx.x * 64;
    const int z  = blockIdx.z;

    const __bf16* Abase = A + (long)z * strideAz + (long)(m0 + l16) * lda;
    const __bf16* Wbase = W + (long)z * strideWz + (long)(n0 + l16) * ldw;

    v8f acc[4][4] = {};

    for (int k0 = 0; k0 < K; k0 += 32) {
        ABfrag a[4];
#pragma unroll
        for (int mi = 0; mi < 4; ++mi) {
            // A fragment: lane<16 holds K=[k0..k0+8)+[k0+16..k0+24); lane>=16 shifted by 8
            const __bf16* pa = Abase + (long)(mi * 16) * lda + k0 + 8 * half;
            a[mi].h[0] = *(const v8bf*)(pa);
            a[mi].h[1] = *(const v8bf*)(pa + 16);
        }
        ABfrag b[4];
#pragma unroll
        for (int ni = 0; ni < 4; ++ni) {
            // B fragment: lane<16 holds K=[k0..k0+8)+[k0+16..k0+24); lane>=16 shifted by 8
            const __bf16* pb = Wbase + (long)(ni * 16) * ldw + k0 + 8 * half;
            b[ni].h[0] = *(const v8bf*)(pb);
            b[ni].h[1] = *(const v8bf*)(pb + 16);
        }
#pragma unroll
        for (int ni = 0; ni < 4; ++ni)
#pragma unroll
            for (int mi = 0; mi < 4; ++mi)
                acc[mi][ni] = __builtin_amdgcn_wmma_f32_16x16x32_bf16(
                    false, a[mi].v, false, b[ni].v, (short)0, acc[mi][ni], false, false);
    }

#pragma unroll
    for (int ni = 0; ni < 4; ++ni) {
        const int n = n0 + ni * 16 + l16;
        const float bn = bias ? bias[n] : 0.0f;
#pragma unroll
        for (int mi = 0; mi < 4; ++mi) {
#pragma unroll
            for (int r = 0; r < 8; ++r) {
                const int m = m0 + mi * 16 + r + 8 * half;
                const float val = acc[mi][ni][r] + bn;
                if constexpr (MODE == 0) {
                    ((float*)outp)[(long)m * ldc + n] = val;
                } else if constexpr (MODE == 2) {
                    ((__bf16*)outp)[(long)z * strideOz + (long)m * ldc + n] = (__bf16)val;
                } else if constexpr (MODE == 1) {
                    const int s  = m / B_DIM, bb = m % B_DIM;
                    const int hh = n / HD_DIM, d = n % HD_DIM;
                    ((__bf16*)outp)[((long)(bb * H_DIM + hh) * HD_DIM + d) * S_DIM + s] = (__bf16)val;
                } else { // MODE 3
                    const int bb = z / H_DIM, hh = z % H_DIM;
                    ((__bf16*)outp)[((long)m * B_DIM + bb) * E_DIM + hh * HD_DIM + n] = (__bf16)val;
                }
            }
        }
    }
}

// ---------------- RoPE (+ optional q scale), fp32 -> bf16 (b,h,tok,d) ----------------
__global__ void rope_kernel(const float* __restrict__ proj,   // rows tok*B+b, cols E
                            const float* __restrict__ pos,    // (B, ntok, E, 2)
                            __bf16* __restrict__ outp,        // (b, h, tok, d)
                            int ntok, float scale) {
    const long npairs = (long)ntok * B_DIM * (E_DIM / 2);
    long p = (long)blockIdx.x * blockDim.x + threadIdx.x;
    if (p >= npairs) return;
    const int  i   = (int)(p % (E_DIM / 2));
    const long m   = p / (E_DIM / 2);
    const int  bb  = (int)(m % B_DIM);
    const int  tok = (int)(m / B_DIM);
    const int  e   = 2 * i;

    const float2 x  = *(const float2*)(proj + m * E_DIM + e);
    const float4 cs = *(const float4*)(pos + (((long)bb * ntok + tok) * E_DIM + e) * 2);
    const float x0 = x.x * scale, x1 = x.y * scale;
    const float y0 = x0 * cs.x - x1 * cs.y;   // even: x0*cos - x1*sin
    const float y1 = x1 * cs.z + x0 * cs.w;   // odd : x1*cos + x0*sin

    const int hh = e / HD_DIM, d = e % HD_DIM;
    __bf16* o = outp + (((long)(bb * H_DIM + hh) * ntok + tok) * HD_DIM + d);
    o[0] = (__bf16)y0;
    o[1] = (__bf16)y1;
}

// ---------------- masked softmax over S, in-place on bf16 weights ----------------
__global__ void softmax_kernel(__bf16* __restrict__ wts, const uint8_t* __restrict__ mask) {
    __shared__ float sdata[256];
    const int  tid = threadIdx.x;
    const long row = blockIdx.x;              // (b*H + h)*T + t
    const int  b   = (int)((row / T_DIM) / H_DIM);
    __bf16* wrow = wts + row * S_DIM;
    const uint8_t* mrow = mask + (long)b * S_DIM;

    float vals[8];
    float lmax = -3.0e38f;
#pragma unroll
    for (int j = 0; j < 8; ++j) {
        const int s = tid + j * 256;
        const float v = mrow[s] ? -3.0e38f : (float)wrow[s];
        vals[j] = v;
        lmax = fmaxf(lmax, v);
    }
    sdata[tid] = lmax; __syncthreads();
    for (int st = 128; st > 0; st >>= 1) {
        if (tid < st) sdata[tid] = fmaxf(sdata[tid], sdata[tid + st]);
        __syncthreads();
    }
    const float rmax = sdata[0]; __syncthreads();

    float lsum = 0.0f;
#pragma unroll
    for (int j = 0; j < 8; ++j) {
        const float ev = __expf(vals[j] - rmax);   // masked -> exp(huge neg) = 0
        vals[j] = ev;
        lsum += ev;
    }
    sdata[tid] = lsum; __syncthreads();
    for (int st = 128; st > 0; st >>= 1) {
        if (tid < st) sdata[tid] += sdata[tid + st];
        __syncthreads();
    }
    const float rinv = 1.0f / sdata[0];
#pragma unroll
    for (int j = 0; j < 8; ++j) wrow[tid + j * 256] = (__bf16)(vals[j] * rinv);
}

// ---------------- mean over heads: (B,H,T,S) bf16 -> (B,T,S) fp32 ----------------
__global__ void meanw_kernel(const __bf16* __restrict__ wts, float* __restrict__ outw) {
    const long total = (long)B_DIM * T_DIM * S_DIM;
    long idx = (long)blockIdx.x * blockDim.x + threadIdx.x;
    if (idx >= total) return;
    const int  s  = (int)(idx % S_DIM);
    const long bt = idx / S_DIM;
    const int  t  = (int)(bt % T_DIM);
    const int  b  = (int)(bt / T_DIM);
    float acc = 0.0f;
#pragma unroll
    for (int hh = 0; hh < H_DIM; ++hh)
        acc += (float)wts[((long)(b * H_DIM + hh) * T_DIM + t) * S_DIM + s];
    outw[idx] = acc * (1.0f / H_DIM);
}

// ---------------- residual + LayerNorm ----------------
__global__ void ln_kernel(const float* __restrict__ proj,    // attn@Wo^T + out_b, rows t*B+b
                          const float* __restrict__ query,   // (T,B,E) flat
                          const float* __restrict__ g, const float* __restrict__ be,
                          float* __restrict__ outp) {
    __shared__ float sdata[256];
    const int  tid = threadIdx.x;
    const long row = blockIdx.x;
    const float* prow = proj  + row * E_DIM;
    const float* qrow = query + row * E_DIM;

    float y[4];
    float lsum = 0.0f;
#pragma unroll
    for (int j = 0; j < 4; ++j) {
        const int e = tid + j * 256;
        y[j] = qrow[e] + prow[e];
        lsum += y[j];
    }
    sdata[tid] = lsum; __syncthreads();
    for (int st = 128; st > 0; st >>= 1) {
        if (tid < st) sdata[tid] += sdata[tid + st];
        __syncthreads();
    }
    const float mu = sdata[0] * (1.0f / E_DIM); __syncthreads();

    float lv = 0.0f;
#pragma unroll
    for (int j = 0; j < 4; ++j) { const float d = y[j] - mu; lv += d * d; }
    sdata[tid] = lv; __syncthreads();
    for (int st = 128; st > 0; st >>= 1) {
        if (tid < st) sdata[tid] += sdata[tid + st];
        __syncthreads();
    }
    const float rstd = rsqrtf(sdata[0] * (1.0f / E_DIM) + LN_EPS);
#pragma unroll
    for (int j = 0; j < 4; ++j) {
        const int e = tid + j * 256;
        outp[row * E_DIM + e] = (y[j] - mu) * rstd * g[e] + be[e];
    }
}

// ======================= host side =======================
extern "C" void kernel_launch(void* const* d_in, const int* in_sizes, int n_in,
                              void* d_out, int out_size, void* d_ws, size_t ws_size,
                              hipStream_t stream) {
    const float*   query     = (const float*)d_in[0];
    const float*   value     = (const float*)d_in[1];
    const float*   query_pos = (const float*)d_in[2];
    const float*   value_pos = (const float*)d_in[3];
    const uint8_t* pad_mask  = (const uint8_t*)d_in[4];
    const float*   in_proj_w = (const float*)d_in[5];
    const float*   in_proj_b = (const float*)d_in[6];
    const float*   out_w     = (const float*)d_in[7];
    const float*   out_b     = (const float*)d_in[8];
    const float*   ln_g      = (const float*)d_in[9];
    const float*   ln_b      = (const float*)d_in[10];

    float* out1 = (float*)d_out;                                   // (T,B,E)
    float* out2 = out1 + (long)T_DIM * B_DIM * E_DIM;              // (B,T,S)

    // workspace carve-out (256B aligned)
    char*  ws  = (char*)d_ws;
    size_t off = 0;
    auto carve = [&](size_t bytes) -> char* {
        off = (off + 255) & ~(size_t)255;
        char* p = ws + off;
        off += bytes;
        return p;
    };
    const long nQrows = (long)T_DIM * B_DIM;   // 4096
    const long nVrows = (long)S_DIM * B_DIM;   // 16384
    __bf16* Wb    = (__bf16*)carve(3L * E_DIM * E_DIM * 2);        // in_proj_w bf16
    __bf16* Wo    = (__bf16*)carve((long)E_DIM * E_DIM * 2);       // out_w bf16
    __bf16* Xq    = (__bf16*)carve(nQrows * E_DIM * 2);            // query bf16
    __bf16* Xv    = (__bf16*)carve(nVrows * E_DIM * 2);            // value bf16
    float*  qproj = (float*) carve(nQrows * E_DIM * 4);            // q pre-RoPE (reused for out-proj)
    float*  kproj = (float*) carve(nVrows * E_DIM * 4);            // k pre-RoPE
    __bf16* qb    = (__bf16*)carve((long)B_DIM * H_DIM * T_DIM * HD_DIM * 2);
    __bf16* kb    = (__bf16*)carve((long)B_DIM * H_DIM * S_DIM * HD_DIM * 2);
    __bf16* vt    = (__bf16*)carve((long)B_DIM * H_DIM * HD_DIM * S_DIM * 2);
    __bf16* wts   = (__bf16*)carve((long)B_DIM * H_DIM * T_DIM * S_DIM * 2);
    __bf16* attnb = (__bf16*)carve(nQrows * E_DIM * 2);
    float*  oproj = qproj;                                         // reuse (sequenced)

    const dim3 blk32(32, 1, 1), blk256(256, 1, 1);

    // 1) conversions
    cvt_f32_to_bf16<<<dim3((unsigned)(3L * E_DIM * E_DIM / 1024)), blk256, 0, stream>>>(in_proj_w, Wb, 3L * E_DIM * E_DIM);
    cvt_f32_to_bf16<<<dim3((unsigned)((long)E_DIM * E_DIM / 1024)), blk256, 0, stream>>>(out_w, Wo, (long)E_DIM * E_DIM);
    cvt_f32_to_bf16<<<dim3((unsigned)(nQrows * E_DIM / 1024)), blk256, 0, stream>>>(query, Xq, nQrows * E_DIM);
    cvt_f32_to_bf16<<<dim3((unsigned)(nVrows * E_DIM / 1024)), blk256, 0, stream>>>(value, Xv, nVrows * E_DIM);

    // 2) QKV projections (WMMA, 64x64 macro-tiles)
    wmma_gemm_kernel<0><<<dim3(E_DIM / 64, nQrows / 64, 1), blk32, 0, stream>>>(
        Xq, E_DIM, 0, Wb, E_DIM, 0, in_proj_b, qproj, E_DIM, 0, E_DIM);
    wmma_gemm_kernel<0><<<dim3(E_DIM / 64, nVrows / 64, 1), blk32, 0, stream>>>(
        Xv, E_DIM, 0, Wb + (long)E_DIM * E_DIM, E_DIM, 0, in_proj_b + E_DIM, kproj, E_DIM, 0, E_DIM);
    wmma_gemm_kernel<1><<<dim3(E_DIM / 64, nVrows / 64, 1), blk32, 0, stream>>>(
        Xv, E_DIM, 0, Wb + 2L * E_DIM * E_DIM, E_DIM, 0, in_proj_b + 2 * E_DIM, vt, 0, 0, E_DIM);

    // 3) RoPE (q also picks up HD^-0.5 scale; commutes with rotation)
    rope_kernel<<<dim3((unsigned)(nQrows * (E_DIM / 2) / 256)), blk256, 0, stream>>>(
        qproj, query_pos, qb, T_DIM, 0.08838834764831845f);
    rope_kernel<<<dim3((unsigned)(nVrows * (E_DIM / 2) / 256)), blk256, 0, stream>>>(
        kproj, value_pos, kb, S_DIM, 1.0f);

    // 4) scores = q @ k^T per (b,h)  (WMMA, bf16 output)
    wmma_gemm_kernel<2><<<dim3(S_DIM / 64, T_DIM / 64, B_DIM * H_DIM), blk32, 0, stream>>>(
        qb, HD_DIM, (long)T_DIM * HD_DIM, kb, HD_DIM, (long)S_DIM * HD_DIM, nullptr,
        wts, S_DIM, (long)T_DIM * S_DIM, HD_DIM);

    // 5) masked softmax (in place) + head-mean into output 2
    softmax_kernel<<<dim3(B_DIM * H_DIM * T_DIM), blk256, 0, stream>>>(wts, pad_mask);
    meanw_kernel<<<dim3((unsigned)((long)B_DIM * T_DIM * S_DIM / 256)), blk256, 0, stream>>>(wts, out2);

    // 6) attn = weights @ v per (b,h)  (WMMA, B-operand is vt so K is contiguous)
    wmma_gemm_kernel<3><<<dim3(HD_DIM / 64, T_DIM / 64, B_DIM * H_DIM), blk32, 0, stream>>>(
        wts, S_DIM, (long)T_DIM * S_DIM, vt, S_DIM, (long)HD_DIM * S_DIM, nullptr,
        attnb, 0, 0, S_DIM);

    // 7) out projection (+out_b) (WMMA), then residual + LayerNorm
    wmma_gemm_kernel<0><<<dim3(E_DIM / 64, nQrows / 64, 1), blk32, 0, stream>>>(
        attnb, E_DIM, 0, Wo, E_DIM, 0, out_b, oproj, E_DIM, 0, E_DIM);
    ln_kernel<<<dim3(T_DIM * B_DIM), blk256, 0, stream>>>(oproj, query, ln_g, ln_b, out1);
}
